// TransBlock_68118181315304
// MI455X (gfx1250) — compile-verified, matrix operand order checked by reading, not run on hardware
//
#include <hip/hip_runtime.h>

// ---- problem constants (match reference) ----
#define Bb 4
#define Ss 1024
#define Dd 1280
#define Hh 20
#define DHd 64
#define DFf 5120
#define Mm (Bb * Ss)   // 4096 rows

typedef __bf16 v16bf __attribute__((ext_vector_type(16)));
typedef float  v8f   __attribute__((ext_vector_type(8)));
typedef unsigned int v4u  __attribute__((ext_vector_type(4)));
typedef int          v4i  __attribute__((ext_vector_type(4)));
typedef int          v8i_ __attribute__((ext_vector_type(8)));

// ---- CDNA5 TDM availability (probe-safe) ----
#if defined(__gfx1250__) && __has_builtin(__builtin_amdgcn_tensor_load_to_lds)
#define HAS_TDM 1
#else
#define HAS_TDM 0
#endif

__device__ __forceinline__ void wait_tdm() {
#if HAS_TDM
#if __has_builtin(__builtin_amdgcn_s_wait_tensorcnt)
    __builtin_amdgcn_s_wait_tensorcnt(0);
#else
    asm volatile("s_wait_tensorcnt 0" ::: "memory");
#endif
#endif
}

// Issue a 2D TDM tile load (bf16 elements): tile [tileh rows x tilew cols]
// from a row-major tensor (row stride = stride elements) into LDS at lds_byte.
__device__ __forceinline__ void tdm_load_2d(const void* g, unsigned lds_byte,
                                            unsigned tw, unsigned th,
                                            unsigned tilew, unsigned tileh,
                                            unsigned stride) {
#if HAS_TDM
    unsigned long long ga = (unsigned long long)g;
    v4u g0;
    g0.x = 1u;                                              // count=1 (valid user D#)
    g0.y = lds_byte;                                        // lds_addr (bytes)
    g0.z = (unsigned)(ga & 0xffffffffull);                  // global_addr[31:0]
    g0.w = (unsigned)((ga >> 32) & 0x01ffffffull) | (2u << 30); // global_addr[56:32] | type=2
    unsigned w0 = (1u << 16);                               // data_size=1 (2 bytes), mask=0
    unsigned w1 = (tw & 0xffffu) << 16;                     // tensor_dim0[15:0] @ b63:48
    unsigned w2 = (tw >> 16) | ((th & 0xffffu) << 16);      // tensor_dim0[31:16] | tensor_dim1[15:0]
    unsigned w3 = (th >> 16) | (tilew << 16);               // tensor_dim1[31:16] | tile_dim0
    unsigned w4 = tileh & 0xffffu;                          // tile_dim1 (tile_dim2 = 0)
    unsigned w5 = stride;                                   // tensor_dim0_stride[31:0]
    v8i_ g1 = { (int)w0, (int)w1, (int)w2, (int)w3, (int)w4, (int)w5, 0, 0 };
    v4i gz = { 0, 0, 0, 0 };
#if __clang_major__ >= 23
    v8i_ gz8 = { 0, 0, 0, 0, 0, 0, 0, 0 };
    __builtin_amdgcn_tensor_load_to_lds(g0, g1, gz, gz, gz8, 0);
#else
    __builtin_amdgcn_tensor_load_to_lds(g0, g1, gz, gz, 0);
#endif
#else
    (void)g; (void)lds_byte; (void)tw; (void)th; (void)tilew; (void)tileh; (void)stride;
#endif
}

__device__ __forceinline__ v16bf load_frag(const __bf16* p0, const __bf16* p1) {
    union { v16bf v; float4 f[2]; } u;
    u.f[0] = *reinterpret_cast<const float4*>(p0);
    u.f[1] = *reinterpret_cast<const float4*>(p1);
    return u.v;
}

__device__ __forceinline__ v8f wmma_bf16(v16bf a, v16bf b, v8f c) {
    return __builtin_amdgcn_wmma_f32_16x16x32_bf16(
        /*neg_a=*/false, a, /*neg_b=*/false, b,
        /*c_mod=*/(short)0, c, /*reuse_a=*/false, /*reuse_b=*/false);
}

// ---------------------------------------------------------------------------
// f32 -> bf16 cast
__global__ void cast_bf16_k(const float* __restrict__ in, __bf16* __restrict__ out, size_t n) {
    size_t i = (size_t)blockIdx.x * blockDim.x + threadIdx.x;
    if (i < n) out[i] = (__bf16)in[i];
}

// f32 [K][N] -> bf16 [N][K] (weight transpose so GEMM B-operand is [col][k])
__global__ void transpose_cast_k(const float* __restrict__ in, __bf16* __restrict__ out,
                                 int Kd, int Nd) {
    size_t i = (size_t)blockIdx.x * blockDim.x + threadIdx.x;
    size_t n = (size_t)Kd * Nd;
    if (i < n) {
        int kk = (int)(i / Nd);
        int nn = (int)(i % Nd);
        out[(size_t)nn * Kd + kk] = (__bf16)in[i];
    }
}

// ---------------------------------------------------------------------------
// Generic bf16 GEMM: C[M][N] = act(scale * A[M][K] @ Bt[N][K]^T + bias) (+resid)
// TDM-staged, LDS double-buffered.
#define BM 128
#define BN 128
#define BK 32

__global__ __launch_bounds__(256) void gemm_bf16_k(
    const __bf16* __restrict__ A, const __bf16* __restrict__ Bt,
    const float* __restrict__ bias, const float* __restrict__ resid,
    float* __restrict__ Cf, __bf16* __restrict__ Cb,
    int M, int N, int K, float scale, int do_relu)
{
    __shared__ __bf16 smem[2 * (BM * BK + BN * BK)];   // 2 x (8KB A + 8KB B)

    const int tid = threadIdx.x;
    const int wid = tid >> 5;
    const int lane = tid & 31;
    const int hl = lane >> 4;        // 16-lane half
    const int l16 = lane & 15;

    const int bm = blockIdx.x * BM;
    const int bn = blockIdx.y * BN;
    const int waveM = (wid & 3) * 32;   // 4 row-groups of 32
    const int waveN = (wid >> 2) * 64;  // 2 col-groups of 64

    const unsigned ldsBase = (unsigned)(size_t)(void*)smem;   // LDS aperture: low 32 bits = offset
    const unsigned bufBytes = (BM * BK + BN * BK) * 2;        // 16384

    auto stage = [&](int pbuf, int kk) {
#if HAS_TDM
        if (tid == 0) {
            tdm_load_2d(A + (size_t)bm * K + kk, ldsBase + (unsigned)pbuf * bufBytes,
                        (unsigned)K, (unsigned)M, BK, BM, (unsigned)K);
            tdm_load_2d(Bt + (size_t)bn * K + kk, ldsBase + (unsigned)pbuf * bufBytes + BM * BK * 2,
                        (unsigned)K, (unsigned)N, BK, BN, (unsigned)K);
        }
#else
        __bf16* dA = smem + pbuf * (BM * BK + BN * BK);
        __bf16* dB = dA + BM * BK;
#pragma unroll
        for (int i = 0; i < 2; ++i) {
            int c = tid + i * 256;
            int r = c >> 2, cc = c & 3;
            *reinterpret_cast<float4*>(&dA[r * BK + cc * 8]) =
                *reinterpret_cast<const float4*>(A + (size_t)(bm + r) * K + kk + cc * 8);
        }
#pragma unroll
        for (int i = 0; i < 2; ++i) {
            int c = tid + i * 256;
            int r = c >> 2, cc = c & 3;
            *reinterpret_cast<float4*>(&dB[r * BK + cc * 8]) =
                *reinterpret_cast<const float4*>(Bt + (size_t)(bn + r) * K + kk + cc * 8);
        }
#endif
    };

    v8f acc[2][4];
#pragma unroll
    for (int mi = 0; mi < 2; ++mi)
#pragma unroll
        for (int ni = 0; ni < 4; ++ni) acc[mi][ni] = (v8f){0,0,0,0,0,0,0,0};

    int buf = 0;
    stage(0, 0);
    wait_tdm();
    __syncthreads();

    for (int k0 = 0; k0 < K; k0 += BK) {
        if (k0 + BK < K) stage(buf ^ 1, k0 + BK);   // DMA next tile while computing

        const __bf16* sA = smem + buf * (BM * BK + BN * BK);
        const __bf16* sB = sA + BM * BK;

        v16bf af[2], bfr[4];
#pragma unroll
        for (int mi = 0; mi < 2; ++mi) {
            int r = waveM + mi * 16 + l16;
            af[mi] = load_frag(&sA[r * BK + hl * 8], &sA[r * BK + 16 + hl * 8]);
        }
#pragma unroll
        for (int ni = 0; ni < 4; ++ni) {
            int r = waveN + ni * 16 + l16;
            bfr[ni] = load_frag(&sB[r * BK + hl * 8], &sB[r * BK + 16 + hl * 8]);
        }
#pragma unroll
        for (int mi = 0; mi < 2; ++mi)
#pragma unroll
            for (int ni = 0; ni < 4; ++ni)
                acc[mi][ni] = wmma_bf16(af[mi], bfr[ni], acc[mi][ni]);

        wait_tdm();          // wave 0 waits its outstanding tensor ops, others no-op
        __syncthreads();     // publish buf^1 / protect buf before overwrite
        buf ^= 1;
    }

    // epilogue: C layout VGPR r, lane: row = r + 8*half, col = lane&15
#pragma unroll
    for (int mi = 0; mi < 2; ++mi)
#pragma unroll
        for (int ni = 0; ni < 4; ++ni) {
            int col = bn + waveN + ni * 16 + l16;
            float bv = bias ? bias[col] : 0.0f;
#pragma unroll
            for (int r = 0; r < 8; ++r) {
                int row = bm + waveM + mi * 16 + r + hl * 8;
                float v = acc[mi][ni][r] * scale + bv;
                if (do_relu) v = fmaxf(v, 0.0f);
                if (resid) v += resid[(size_t)row * N + col];
                if (Cf) Cf[(size_t)row * N + col] = v;
                if (Cb) Cb[(size_t)row * N + col] = (__bf16)v;
            }
        }
}

// ---------------------------------------------------------------------------
// Flash attention: per block one (b, h, 64-query tile); 4 waves x 16 rows.
__global__ __launch_bounds__(128) void attn_k(
    const __bf16* __restrict__ q, const __bf16* __restrict__ k,
    const __bf16* __restrict__ v, float* __restrict__ out)
{
    const int qt = blockIdx.x, h = blockIdx.y, b = blockIdx.z;
    __shared__ __bf16 sK[32 * 64];       // [key][dh]
    __shared__ __bf16 sVt[64 * 32];      // [dh][key]
    __shared__ __bf16 sP[4][16 * 32];    // per-wave probs [qrow][key]

    const int tid = threadIdx.x;
    const int wid = tid >> 5;
    const int lane = tid & 31;
    const int hl = lane >> 4;
    const int l16 = lane & 15;
    const int ld = Hh * DHd;             // 1280
    const int qrow0 = qt * 64 + wid * 16;

    // Q fragments (16x64 split into two 16x32 A-frags), pre-scaled 1/sqrt(DH)
    size_t qbase = ((size_t)b * Ss + qrow0 + l16) * ld + h * DHd;
    v16bf qa0 = load_frag(q + qbase + hl * 8,      q + qbase + 16 + hl * 8);
    v16bf qa1 = load_frag(q + qbase + 32 + hl * 8, q + qbase + 48 + hl * 8);

    v8f acc[4];
#pragma unroll
    for (int nt = 0; nt < 4; ++nt) acc[nt] = (v8f){0,0,0,0,0,0,0,0};
    float mrow[8], lrow[8];
#pragma unroll
    for (int r = 0; r < 8; ++r) { mrow[r] = -1e30f; lrow[r] = 0.0f; }

    for (int kt = 0; kt < Ss; kt += 32) {
        // stage K tile [32 keys][64 dh]
#pragma unroll
        for (int i = 0; i < 2; ++i) {
            int c = tid + i * 128;
            int key = c >> 3, cc = c & 7;
            *reinterpret_cast<float4*>(&sK[key * 64 + cc * 8]) =
                *reinterpret_cast<const float4*>(
                    k + ((size_t)b * Ss + kt + key) * ld + h * DHd + cc * 8);
        }
        // stage V tile transposed -> [dh][key]
#pragma unroll
        for (int i = 0; i < 16; ++i) {
            int e = tid * 16 + i;            // 0..2047
            int key = e >> 6, dh = e & 63;
            sVt[dh * 32 + key] = v[((size_t)b * Ss + kt + key) * ld + h * DHd + dh];
        }
        __syncthreads();

        // scores for two 16-key sub-tiles, with online softmax
#pragma unroll
        for (int ns = 0; ns < 2; ++ns) {
            int krow = ns * 16 + l16;
            v16bf kb0 = load_frag(&sK[krow * 64 + hl * 8],      &sK[krow * 64 + 16 + hl * 8]);
            v16bf kb1 = load_frag(&sK[krow * 64 + 32 + hl * 8], &sK[krow * 64 + 48 + hl * 8]);
            v8f s = (v8f){0,0,0,0,0,0,0,0};
            s = wmma_bf16(qa0, kb0, s);
            s = wmma_bf16(qa1, kb1, s);
#pragma unroll
            for (int r = 0; r < 8; ++r) {
                float x = s[r];
#pragma unroll
                for (int m = 1; m < 16; m <<= 1) x = fmaxf(x, __shfl_xor(x, m, 32));
                float mnew = fmaxf(mrow[r], x);
                float corr = __expf(mrow[r] - mnew);
#pragma unroll
                for (int nt = 0; nt < 4; ++nt) acc[nt][r] *= corr;
                float p = __expf(s[r] - mnew);
                float ps = p;
#pragma unroll
                for (int m = 1; m < 16; m <<= 1) ps += __shfl_xor(ps, m, 32);
                lrow[r] = lrow[r] * corr + ps;
                mrow[r] = mnew;
                int prow = r + hl * 8;
                sP[wid][prow * 32 + ns * 16 + l16] = (__bf16)p;
            }
        }
#if __HIP_DEVICE_COMPILE__
        asm volatile("s_wait_dscnt 0" ::: "memory");   // wave-local LDS RAW fence
#endif
        // P (16x32) @ V (32xDH): 4 output N-tiles of 16 dh
        {
            int r = l16;
            v16bf pa = load_frag(&sP[wid][r * 32 + hl * 8], &sP[wid][r * 32 + 16 + hl * 8]);
#pragma unroll
            for (int nt = 0; nt < 4; ++nt) {
                int vrow = nt * 16 + l16;
                v16bf vb = load_frag(&sVt[vrow * 32 + hl * 8], &sVt[vrow * 32 + 16 + hl * 8]);
                acc[nt] = wmma_bf16(pa, vb, acc[nt]);
            }
        }
        __syncthreads();
    }

    // normalize and write f32 attention output [B*S][H*DH]
#pragma unroll
    for (int nt = 0; nt < 4; ++nt)
#pragma unroll
        for (int r = 0; r < 8; ++r) {
            int row = qrow0 + r + hl * 8;
            int col = h * DHd + nt * 16 + l16;
            out[((size_t)b * Ss + row) * ld + col] = acc[nt][r] / lrow[r];
        }
}

// ---------------------------------------------------------------------------
// residual add + LayerNorm -> f32 residual copy + bf16 normalized activations
__global__ __launch_bounds__(256) void resid_ln_k(
    const float* __restrict__ x, const float* __restrict__ attn,
    const float* __restrict__ g, const float* __restrict__ bta,
    float* __restrict__ xr, __bf16* __restrict__ hb)
{
    const int row = blockIdx.x;
    __shared__ float red[8];
    size_t base = (size_t)row * Dd;
    float vals[5];
    float s = 0.0f;
#pragma unroll
    for (int i = 0; i < 5; ++i) {
        int c = threadIdx.x + i * 256;
        float t = x[base + c] + attn[base + c];
        vals[i] = t; s += t;
        xr[base + c] = t;
    }
#pragma unroll
    for (int m = 1; m < 32; m <<= 1) s += __shfl_xor(s, m, 32);
    if ((threadIdx.x & 31) == 0) red[threadIdx.x >> 5] = s;
    __syncthreads();
    float tot = 0.0f;
#pragma unroll
    for (int w = 0; w < 8; ++w) tot += red[w];
    float mu = tot / (float)Dd;

    float vs = 0.0f;
#pragma unroll
    for (int i = 0; i < 5; ++i) { float d = vals[i] - mu; vs += d * d; }
#pragma unroll
    for (int m = 1; m < 32; m <<= 1) vs += __shfl_xor(vs, m, 32);
    __syncthreads();
    if ((threadIdx.x & 31) == 0) red[threadIdx.x >> 5] = vs;
    __syncthreads();
    float vtot = 0.0f;
#pragma unroll
    for (int w = 0; w < 8; ++w) vtot += red[w];
    float rstd = rsqrtf(vtot / (float)Dd + 1e-5f);

#pragma unroll
    for (int i = 0; i < 5; ++i) {
        int c = threadIdx.x + i * 256;
        hb[base + c] = (__bf16)((vals[i] - mu) * rstd * g[c] + bta[c]);
    }
}

// ---------------------------------------------------------------------------
extern "C" void kernel_launch(void* const* d_in, const int* in_sizes, int n_in,
                              void* d_out, int out_size, void* d_ws, size_t ws_size,
                              hipStream_t stream) {
    const float* x    = (const float*)d_in[0];
    const float* Wq   = (const float*)d_in[1];
    const float* Wk   = (const float*)d_in[2];
    const float* Wv   = (const float*)d_in[3];
    const float* ln_g = (const float*)d_in[4];
    const float* ln_b = (const float*)d_in[5];
    const float* W1   = (const float*)d_in[6];
    const float* b1   = (const float*)d_in[7];
    const float* W2   = (const float*)d_in[8];
    const float* b2   = (const float*)d_in[9];
    float* out = (float*)d_out;

    char* ws = (char*)d_ws;
    size_t o = 0;
    auto alloc = [&](size_t bytes) { char* p = ws + o; o += (bytes + 255) & ~(size_t)255; return p; };
    __bf16* xb   = (__bf16*)alloc((size_t)Mm * Dd * 2);
    __bf16* wqt  = (__bf16*)alloc((size_t)Dd * Dd * 2);
    __bf16* wkt  = (__bf16*)alloc((size_t)Dd * Dd * 2);
    __bf16* wvt  = (__bf16*)alloc((size_t)Dd * Dd * 2);
    __bf16* w1t  = (__bf16*)alloc((size_t)Dd * DFf * 2);
    __bf16* w2t  = (__bf16*)alloc((size_t)DFf * Dd * 2);
    __bf16* qb   = (__bf16*)alloc((size_t)Mm * Dd * 2);
    __bf16* kb   = (__bf16*)alloc((size_t)Mm * Dd * 2);
    __bf16* vb   = (__bf16*)alloc((size_t)Mm * Dd * 2);
    float*  attn = (float*)alloc((size_t)Mm * Dd * 4);
    float*  xr   = (float*)alloc((size_t)Mm * Dd * 4);
    __bf16* hb   = (__bf16*)alloc((size_t)Mm * Dd * 2);
    __bf16* hh   = (__bf16*)alloc((size_t)Mm * DFf * 2);
    (void)ws_size; (void)in_sizes; (void)n_in; (void)out_size;

    // 1) casts / weight transposes
    {
        size_t n = (size_t)Mm * Dd;
        cast_bf16_k<<<dim3((unsigned)((n + 255) / 256)), dim3(256), 0, stream>>>(x, xb, n);
        size_t nw = (size_t)Dd * Dd;
        transpose_cast_k<<<dim3((unsigned)((nw + 255) / 256)), dim3(256), 0, stream>>>(Wq, wqt, Dd, Dd);
        transpose_cast_k<<<dim3((unsigned)((nw + 255) / 256)), dim3(256), 0, stream>>>(Wk, wkt, Dd, Dd);
        transpose_cast_k<<<dim3((unsigned)((nw + 255) / 256)), dim3(256), 0, stream>>>(Wv, wvt, Dd, Dd);
        size_t n1 = (size_t)Dd * DFf;
        transpose_cast_k<<<dim3((unsigned)((n1 + 255) / 256)), dim3(256), 0, stream>>>(W1, w1t, Dd, DFf);
        transpose_cast_k<<<dim3((unsigned)((n1 + 255) / 256)), dim3(256), 0, stream>>>(W2, w2t, DFf, Dd);
    }

    // 2) QKV projections (q pre-scaled by 1/sqrt(DH))
    {
        dim3 g(Mm / BM, Dd / BN);
        gemm_bf16_k<<<g, 256, 0, stream>>>(xb, wqt, nullptr, nullptr, nullptr, qb,
                                           Mm, Dd, Dd, 0.125f, 0);
        gemm_bf16_k<<<g, 256, 0, stream>>>(xb, wkt, nullptr, nullptr, nullptr, kb,
                                           Mm, Dd, Dd, 1.0f, 0);
        gemm_bf16_k<<<g, 256, 0, stream>>>(xb, wvt, nullptr, nullptr, nullptr, vb,
                                           Mm, Dd, Dd, 1.0f, 0);
    }

    // 3) flash attention
    attn_k<<<dim3(Ss / 64, Hh, Bb), 128, 0, stream>>>(qb, kb, vb, attn);

    // 4) residual + LayerNorm
    resid_ln_k<<<dim3(Mm), 256, 0, stream>>>(x, attn, ln_g, ln_b, xr, hb);

    // 5) FFN
    gemm_bf16_k<<<dim3(Mm / BM, DFf / BN), 256, 0, stream>>>(
        hb, w1t, b1, nullptr, nullptr, hh, Mm, DFf, Dd, 1.0f, 1 /*relu*/);
    gemm_bf16_k<<<dim3(Mm / BM, Dd / BN), 256, 0, stream>>>(
        hh, w2t, b2, xr, out, nullptr, Mm, Dd, DFf, 1.0f, 0);
}
